// SpectreViT_29334626632033
// MI455X (gfx1250) — compile-verified, matrix operand order checked by reading, not run on hardware
//
#include <hip/hip_runtime.h>

// ---------------------------------------------------------------------------
// SpectreViT on MI455X (gfx1250): everything reduced to bf16 WMMA GEMMs.
//  - rfft2 patch embed folded into a single 768x768 projection matrix
//  - fft2(...).real token mixing = Ch@X@Cw - Sh@X@Sw (4 dense GEMMs)
//  - FF chain = 3 GEMMs/layer; LN fused with residuals
// fp32 in HBM, converted to bf16 with the hardware v_cvt_pk_bf16_f32 during
// LDS staging (weights are L2-resident, 57MB/layer << 192MB L2), fp32
// accumulate via v_wmma_f32_16x16x32_bf16.
// GEMM core: 128x128 block tile, 8 wave32 waves (4x2), 32x64 per wave
// (2x4 WMMA frags), double-buffered LDS, branchless clamped staging,
// K-edge guards only in the template instantiation that needs them (K=197).
// ---------------------------------------------------------------------------

#define P_    16
#define PW_   9
#define D_    768
#define FF_   3072
#define L_    12
#define NCLS_ 1000
#define B_    16
#define C_    3
#define H_    224
#define N_    196
#define S_    197
#define BS_   (B_ * S_)   // 3152 token rows
#define BN_   (B_ * N_)   // 3136 patch rows
#define PPC_  768         // C*P*P pixels per patch
#define FIN_  432         // C*P*PW rfft coeffs per patch
#define LDA_S 208         // padded row stride for Ch/Sh (16B aligned rows)

typedef __bf16 bf16_t;
typedef __attribute__((ext_vector_type(16))) __bf16 v16bf;
typedef __attribute__((ext_vector_type(8)))  float  v8f;

// hardware conversion: lowers to v_cvt_pk_bf16_f32 (RNE per ISA) instead of a
// 3-op integer emulation sequence per element.
__device__ __forceinline__ bf16_t f2bf(float f) { return (bf16_t)f; }

__device__ __forceinline__ int imin(int a, int b) { return a < b ? a : b; }

enum { EPI_STORE = 0, EPI_BIAS = 1, EPI_SUB = 2 };
enum { B_NXK = 0, B_KXN = 1 };

#define TM 128
#define TN 128
#define KB 32
#define KP 40   // padded LDS row stride (bf16 elems): conflict-free frag reads

// C[m,n] (+epilogue) = sum_k A[m,k] * B[.,.]  ; batched via gridDim.z
// GK: K is not a multiple of KB -> per-element guards in staging (only the
//     197-length S-axis DFT GEMMs instantiate this).
template <int EPI, int BL, bool GK>
__global__ __launch_bounds__(256) void gemm_wmma(
    int M, int N, int K,
    const float* __restrict__ A,  long lda, long sA,
    const float* __restrict__ Bm, long ldb, long sB,
    float* __restrict__ Cm,       long ldc, long sC,
    const float* __restrict__ bias)
{
  __shared__ bf16_t As[2][TM * KP];
  __shared__ bf16_t Bs[2][TN * KP];
  const int z = blockIdx.z;
  A  += (long)z * sA;
  Bm += (long)z * sB;
  Cm += (long)z * sC;
  const int m0 = blockIdx.y * TM, n0 = blockIdx.x * TN;
  const int tid  = threadIdx.x;
  const int lane = tid & 31, wid = tid >> 5;
  const int wm = wid >> 1, wn = wid & 1;       // 4x2 waves -> 32x64 per wave
  const int half = lane >> 4, l = lane & 15;

  v8f acc[2][4];
#pragma unroll
  for (int i = 0; i < 2; ++i)
#pragma unroll
    for (int j = 0; j < 4; ++j)
#pragma unroll
      for (int r = 0; r < 8; ++r) acc[i][j][r] = 0.0f;

  // branchless clamped loader coordinates
  const int ar = tid >> 3, ac = (tid & 7) << 2;    // A / B(NxK): 32 rows x 32 cols per pass
  const int bk = tid & 31, bn = (tid >> 5) << 4;   // B(KxN): row k=bk, 16-col group

  float4 ra[4], rb[4];

  // ---- staging: global -> registers (branchless; rows clamped in-bounds) ----
  auto loadA = [&](int k0) {
#pragma unroll
    for (int rr = 0; rr < 4; ++rr) {
      const int m = imin(m0 + ar + rr * 32, M - 1);
      const float* p = A + (long)m * lda;
      if constexpr (!GK) {
        ra[rr] = *(const float4*)(p + k0 + ac);
      } else {
        const int kg = k0 + ac;
        ra[rr].x = (kg + 0 < K) ? p[kg + 0] : 0.f;
        ra[rr].y = (kg + 1 < K) ? p[kg + 1] : 0.f;
        ra[rr].z = (kg + 2 < K) ? p[kg + 2] : 0.f;
        ra[rr].w = (kg + 3 < K) ? p[kg + 3] : 0.f;
      }
    }
  };
  auto loadB = [&](int k0) {
    if constexpr (BL == B_NXK) {
#pragma unroll
      for (int rr = 0; rr < 4; ++rr) {
        const int n = imin(n0 + ar + rr * 32, N - 1);
        const float* p = Bm + (long)n * ldb;
        if constexpr (!GK) {
          rb[rr] = *(const float4*)(p + k0 + ac);
        } else {
          const int kg = k0 + ac;
          rb[rr].x = (kg + 0 < K) ? p[kg + 0] : 0.f;
          rb[rr].y = (kg + 1 < K) ? p[kg + 1] : 0.f;
          rb[rr].z = (kg + 2 < K) ? p[kg + 2] : 0.f;
          rb[rr].w = (kg + 3 < K) ? p[kg + 3] : 0.f;
        }
      }
    } else {  // B stored [K][N] row-major (call sites guarantee N % TN == 0)
      const int kg = k0 + bk;
      const int kc = GK ? imin(kg, K - 1) : kg;
      const float* p = Bm + (long)kc * ldb + n0 + bn;
#pragma unroll
      for (int rr = 0; rr < 4; ++rr) {
        float4 t = *(const float4*)(p + rr * 4);
        if constexpr (GK) {
          if (kg >= K) { t.x = 0.f; t.y = 0.f; t.z = 0.f; t.w = 0.f; }
        }
        rb[rr] = t;
      }
    }
  };
  // ---- staging: registers -> LDS (bf16, packed cvt) ----
  auto storeA = [&](int buf) {
#pragma unroll
    for (int rr = 0; rr < 4; ++rr) {
      bf16_t* d = &As[buf][(ar + rr * 32) * KP + ac];
      d[0] = f2bf(ra[rr].x); d[1] = f2bf(ra[rr].y);
      d[2] = f2bf(ra[rr].z); d[3] = f2bf(ra[rr].w);
    }
  };
  auto storeB = [&](int buf) {
    if constexpr (BL == B_NXK) {
#pragma unroll
      for (int rr = 0; rr < 4; ++rr) {
        bf16_t* d = &Bs[buf][(ar + rr * 32) * KP + ac];
        d[0] = f2bf(rb[rr].x); d[1] = f2bf(rb[rr].y);
        d[2] = f2bf(rb[rr].z); d[3] = f2bf(rb[rr].w);
      }
    } else {  // transpose while storing: Bs[n][k]
#pragma unroll
      for (int rr = 0; rr < 4; ++rr) {
        Bs[buf][(bn + rr * 4 + 0) * KP + bk] = f2bf(rb[rr].x);
        Bs[buf][(bn + rr * 4 + 1) * KP + bk] = f2bf(rb[rr].y);
        Bs[buf][(bn + rr * 4 + 2) * KP + bk] = f2bf(rb[rr].z);
        Bs[buf][(bn + rr * 4 + 3) * KP + bk] = f2bf(rb[rr].w);
      }
    }
  };
  // ---- compute: LDS fragments -> 8x WMMA ----
  auto compute = [&](int buf) {
    v16bf af[2], bfv[4];
#pragma unroll
    for (int i = 0; i < 2; ++i) {
      const bf16_t* p = &As[buf][(wm * 32 + i * 16 + l) * KP + half * 8];
#pragma unroll
      for (int e = 0; e < 8; ++e) { af[i][e] = p[e]; af[i][e + 8] = p[e + 16]; }
    }
#pragma unroll
    for (int j = 0; j < 4; ++j) {
      const bf16_t* p = &Bs[buf][(wn * 64 + j * 16 + l) * KP + half * 8];
#pragma unroll
      for (int e = 0; e < 8; ++e) { bfv[j][e] = p[e]; bfv[j][e + 8] = p[e + 16]; }
    }
#pragma unroll
    for (int i = 0; i < 2; ++i)
#pragma unroll
      for (int j = 0; j < 4; ++j)
        acc[i][j] = __builtin_amdgcn_wmma_f32_16x16x32_bf16(
            false, af[i], false, bfv[j], (short)0, acc[i][j], false, false);
  };

  // ---- double-buffered main loop: 1 barrier per K-step ----
  const int nk = (K + KB - 1) / KB;
  loadA(0); loadB(0);
  storeA(0); storeB(0);
  int buf = 0;
  for (int kb = 0; kb < nk; ++kb) {
    __syncthreads();
    const bool more = (kb + 1) < nk;
    if (more) { loadA((kb + 1) * KB); loadB((kb + 1) * KB); }
    compute(buf);
    if (more) { storeA(buf ^ 1); storeB(buf ^ 1); buf ^= 1; }
  }

  // ---- epilogue (32-bit C/D 16x16 layout: m = r + 8*half, n = lane&15) ----
#pragma unroll
  for (int i = 0; i < 2; ++i)
#pragma unroll
    for (int j = 0; j < 4; ++j)
#pragma unroll
      for (int r = 0; r < 8; ++r) {
        const int m = m0 + wm * 32 + i * 16 + half * 8 + r;
        const int n = n0 + wn * 64 + j * 16 + l;
        if (m < M && n < N) {
          float v = acc[i][j][r];
          float* cp = Cm + (long)m * ldc + n;
          if (EPI == EPI_BIAS) v += bias[n];
          if (EPI == EPI_SUB)  v = *cp - v;
          *cp = v;
        }
      }
}

// ---------------- LayerNorm (row = 768), fused residual ----------------
// MODE 0: out = LN(A)*g + be + Bv        (post-mix residual)
// MODE 1: out = LN(A + Bv)*g + be        (post-FF norm)
template <int MODE>
__global__ __launch_bounds__(256) void ln_kernel(
    const float* __restrict__ A, const float* __restrict__ Bv,
    const float* __restrict__ g, const float* __restrict__ be,
    float* __restrict__ O)
{
  const int r = blockIdx.x;
  const int tid = threadIdx.x;
  const long base = (long)r * D_;
  float x[3];
  float s = 0.f, s2 = 0.f;
#pragma unroll
  for (int e = 0; e < 3; ++e) {
    const int c = tid + e * 256;
    float v = A[base + c];
    if (MODE == 1) v += Bv[base + c];
    x[e] = v; s += v; s2 += v * v;
  }
  __shared__ float rs[256], rq[256];
  rs[tid] = s; rq[tid] = s2;
  __syncthreads();
  for (int off = 128; off > 0; off >>= 1) {
    if (tid < off) { rs[tid] += rs[tid + off]; rq[tid] += rq[tid + off]; }
    __syncthreads();
  }
  const float mean = rs[0] * (1.0f / 768.0f);
  const float var  = rq[0] * (1.0f / 768.0f) - mean * mean;
  const float inv  = rsqrtf(var + 1e-5f);
#pragma unroll
  for (int e = 0; e < 3; ++e) {
    const int c = tid + e * 256;
    float y = (x[e] - mean) * inv * g[c] + be[c];
    if (MODE == 0) y += Bv[base + c];
    O[base + c] = y;
  }
}

// ---------------- precompute: DFT matrices ----------------
__global__ void build_cwsw(float* __restrict__ CW, float* __restrict__ SW) {
  const int idx = blockIdx.x * 256 + threadIdx.x;
  if (idx >= D_ * D_) return;
  const int q = idx / D_, d = idx % D_;
  const int t = (int)(((long)q * d) % D_);
  const float ang = 6.283185307179586f * (float)t / (float)D_;
  CW[idx] = __cosf(ang);
  SW[idx] = __sinf(ang);
}

__global__ void build_chsh(float* __restrict__ CH, float* __restrict__ SH) {
  const int idx = blockIdx.x * 256 + threadIdx.x;
  if (idx >= S_ * LDA_S) return;
  const int p = idx / LDA_S, s = idx % LDA_S;
  float cv = 0.f, sv = 0.f;
  if (s < S_) {
    const int t = (p * s) % S_;
    const float ang = 6.283185307179586f * (float)t / (float)S_;
    cv = __cosf(ang); sv = __sinf(ang);
  }
  CH[idx] = cv; SH[idx] = sv;
}

// fold rfft2(ortho).real * freq through proj_w -> MF[d][(c,h,w)]
__global__ void build_mfold(const float* __restrict__ pw,
                            const float* __restrict__ fh,
                            const float* __restrict__ fw,
                            float* __restrict__ MF) {
  const long idx = (long)blockIdx.x * 256 + threadIdx.x;
  if (idx >= (long)D_ * PPC_) return;
  const int rcol = (int)(idx % PPC_);
  const int d    = (int)(idx / PPC_);
  const int c = rcol >> 8, hw = rcol & 255, h = hw >> 4, w = hw & 15;
  float ct[16];
#pragma unroll
  for (int t = 0; t < 16; ++t) ct[t] = __cosf(0.39269908169872414f * (float)t);
  const float* pr = pw + (long)d * FIN_ + c * (P_ * PW_);
  float sum = 0.f;
  for (int u = 0; u < P_; ++u) {
    const float fu = fh[u];
#pragma unroll
    for (int v = 0; v < PW_; ++v) {
      const int t = (u * h + v * w) & 15;
      sum += pr[u * PW_ + v] * fu * fw[v] * ct[t];
    }
  }
  MF[idx] = sum * (1.0f / 16.0f);
}

// gather non-overlapping 16x16 patches: XP[b*N+n][(c,h,w)]
__global__ void patch_extract(const float* __restrict__ x, float* __restrict__ XP) {
  const long idx = (long)blockIdx.x * 256 + threadIdx.x;
  if (idx >= (long)BN_ * PPC_) return;
  const int col = (int)(idx % PPC_);
  const long row = idx / PPC_;
  const int b = (int)(row / N_), n = (int)(row % N_);
  const int c = col >> 8, hw = col & 255, h = hw >> 4, w = hw & 15;
  const int ph = n / 14, pw = n % 14;
  XP[idx] = x[((long)(b * C_ + c) * H_ + ph * P_ + h) * H_ + pw * P_ + w];
}

// X[b][s][:] = (s==0 ? cls : xe[b][s-1]) + pos[s]; SRC = copy for final residual
__global__ void assemble(const float* __restrict__ XE, const float* __restrict__ cls,
                         const float* __restrict__ pos,
                         float* __restrict__ X, float* __restrict__ SRC) {
  const long idx = (long)blockIdx.x * 256 + threadIdx.x;
  if (idx >= (long)BS_ * D_) return;
  const int d = (int)(idx % D_);
  const long row = idx / D_;
  const int b = (int)(row / S_), s = (int)(row % S_);
  float v = (s == 0) ? cls[d] : XE[(long)(b * N_ + s - 1) * D_ + d];
  v += pos[(long)s * D_ + d];
  X[idx] = v; SRC[idx] = v;
}

__global__ void add_src(float* __restrict__ X, const float* __restrict__ SRC) {
  const long idx = (long)blockIdx.x * 256 + threadIdx.x;
  if (idx < (long)BS_ * D_) X[idx] += SRC[idx];
}

static inline dim3 gemm_grid(int M, int N, int z) {
  return dim3((unsigned)((N + TN - 1) / TN), (unsigned)((M + TM - 1) / TM), (unsigned)z);
}

extern "C" void kernel_launch(void* const* d_in, const int* in_sizes, int n_in,
                              void* d_out, int out_size, void* d_ws, size_t ws_size,
                              hipStream_t stream) {
  (void)in_sizes; (void)n_in; (void)out_size; (void)ws_size;
  const float* x       = (const float*)d_in[0];
  const float* freq_h  = (const float*)d_in[1];
  const float* freq_w  = (const float*)d_in[2];
  const float* proj_w  = (const float*)d_in[3];
  const float* proj_b  = (const float*)d_in[4];
  const float* cls_tok = (const float*)d_in[5];
  const float* pos_emb = (const float*)d_in[6];
  const float* w1      = (const float*)d_in[7];
  const float* b1      = (const float*)d_in[8];
  const float* w2      = (const float*)d_in[9];
  const float* b2      = (const float*)d_in[10];
  const float* w3      = (const float*)d_in[11];
  const float* b3      = (const float*)d_in[12];
  const float* ln1_g   = (const float*)d_in[13];
  const float* ln1_b   = (const float*)d_in[14];
  const float* ln2_g   = (const float*)d_in[15];
  const float* ln2_b   = (const float*)d_in[16];
  const float* head_w  = (const float*)d_in[17];
  const float* head_b  = (const float*)d_in[18];
  float* out = (float*)d_out;

  const long nXD = (long)BS_ * D_;
  const long nXF = (long)BS_ * FF_;
  float* ws  = (float*)d_ws;
  float* X   = ws; ws += nXD;
  float* SRC = ws; ws += nXD;
  float* T1  = ws; ws += nXD;
  float* T2  = ws; ws += nXD;
  float* MIX = ws; ws += nXD;
  float* O1  = ws; ws += nXD;
  float* F1  = ws; ws += nXF;
  float* F2  = ws; ws += nXF;
  float* XP  = ws; ws += (long)BN_ * PPC_;
  float* XE  = ws; ws += (long)BN_ * D_;
  float* MF  = ws; ws += (long)D_ * PPC_;
  float* CW  = ws; ws += (long)D_ * D_;
  float* SW  = ws; ws += (long)D_ * D_;
  float* CH  = ws; ws += (long)S_ * LDA_S;
  float* SH  = ws; ws += (long)S_ * LDA_S;

  const int T = 256;
  const long sSD = (long)S_ * D_;

  // ---- precompute DFT matrices + folded patch projection ----
  build_cwsw<<<(D_ * D_ + T - 1) / T, T, 0, stream>>>(CW, SW);
  build_chsh<<<(S_ * LDA_S + T - 1) / T, T, 0, stream>>>(CH, SH);
  build_mfold<<<(int)(((long)D_ * PPC_ + T - 1) / T), T, 0, stream>>>(proj_w, freq_h, freq_w, MF);

  // ---- patch embed: gather, then one big GEMM ----
  patch_extract<<<(int)(((long)BN_ * PPC_ + T - 1) / T), T, 0, stream>>>(x, XP);
  gemm_wmma<EPI_BIAS, B_NXK, false><<<gemm_grid(BN_, D_, 1), 256, 0, stream>>>(
      BN_, D_, PPC_, XP, PPC_, 0, MF, PPC_, 0, XE, D_, 0, proj_b);
  assemble<<<(int)((nXD + T - 1) / T), T, 0, stream>>>(XE, cls_tok, pos_emb, X, SRC);

  // ---- 12 encoder layers ----
  for (int i = 0; i < L_; ++i) {
    // D-axis DFT (flat over all tokens): T1 = X @ Cw, T2 = X @ Sw
    gemm_wmma<EPI_STORE, B_NXK, false><<<gemm_grid(BS_, D_, 1), 256, 0, stream>>>(
        BS_, D_, D_, X, D_, 0, CW, D_, 0, T1, D_, 0, nullptr);
    gemm_wmma<EPI_STORE, B_NXK, false><<<gemm_grid(BS_, D_, 1), 256, 0, stream>>>(
        BS_, D_, D_, X, D_, 0, SW, D_, 0, T2, D_, 0, nullptr);
    // S-axis DFT (batched per image, K=197 -> guarded staging): MIX = Ch@T1 - Sh@T2
    gemm_wmma<EPI_STORE, B_KXN, true><<<gemm_grid(S_, D_, B_), 256, 0, stream>>>(
        S_, D_, S_, CH, LDA_S, 0, T1, D_, sSD, MIX, D_, sSD, nullptr);
    gemm_wmma<EPI_SUB, B_KXN, true><<<gemm_grid(S_, D_, B_), 256, 0, stream>>>(
        S_, D_, S_, SH, LDA_S, 0, T2, D_, sSD, MIX, D_, sSD, nullptr);
    // O1 = LN(MIX)*g+b + X
    ln_kernel<0><<<BS_, 256, 0, stream>>>(MIX, X, ln1_g + i * D_, ln1_b + i * D_, O1);
    // FF chain (no activation): F1 = O1@w1^T+b1 ; F2 = F1@w2^T+b2 ; T1 = F2@w3^T+b3
    gemm_wmma<EPI_BIAS, B_NXK, false><<<gemm_grid(BS_, FF_, 1), 256, 0, stream>>>(
        BS_, FF_, D_, O1, D_, 0, w1 + (long)i * FF_ * D_, D_, 0, F1, FF_, 0, b1 + i * FF_);
    gemm_wmma<EPI_BIAS, B_NXK, false><<<gemm_grid(BS_, FF_, 1), 256, 0, stream>>>(
        BS_, FF_, FF_, F1, FF_, 0, w2 + (long)i * FF_ * FF_, FF_, 0, F2, FF_, 0, b2 + i * FF_);
    gemm_wmma<EPI_BIAS, B_NXK, false><<<gemm_grid(BS_, D_, 1), 256, 0, stream>>>(
        BS_, D_, FF_, F2, FF_, 0, w3 + (long)i * D_ * FF_, FF_, 0, T1, D_, 0, b3 + i * D_);
    // X = LN(O1 + T1)*g+b
    ln_kernel<1><<<BS_, 256, 0, stream>>>(O1, T1, ln2_g + i * D_, ln2_b + i * D_, X);
  }

  // ---- encoder-level residual + head on cls tokens ----
  add_src<<<(int)((nXD + T - 1) / T), T, 0, stream>>>(X, SRC);
  gemm_wmma<EPI_BIAS, B_NXK, false><<<gemm_grid(B_, NCLS_, 1), 256, 0, stream>>>(
      B_, NCLS_, D_, X, sSD, 0, head_w, D_, 0, out, NCLS_, 0, head_b);
}
